// convlstm_remnet_cb_91061896609912
// MI455X (gfx1250) — compile-verified, matrix operand order checked by reading, not run on hardware
//
#include <hip/hip_runtime.h>

// ---------------------------------------------------------------------------
// ConvLSTM + memory-guided attention (remnet) for MI455X / gfx1250 (wave32).
// 3x3 gate convs and the 1x1 output conv run as implicit GEMMs on
// v_wmma_f32_16x16x32_f16 with pre-swizzled per-lane fragments.
// A-tiles are DMA'd into LDS with GLOBAL_LOAD_ASYNC_TO_LDS_B128 (ASYNCcnt),
// double-buffered so the async copy overlaps the WMMA K-loop, and shared by
// all 8 waves of the block via ds_load. 2 M-tiles per block double B reuse.
// ---------------------------------------------------------------------------

typedef __attribute__((ext_vector_type(16))) _Float16 v16h;
typedef __attribute__((ext_vector_type(8)))  float    v8f;

#define SEQ    10
#define BATCH  64
#define IN_CH  8
#define HID    64
#define MG     60
#define RLEN   4
#define HW     256                  // 16*16 pixels per image
#define NPIX   (BATCH * HW)         // 16384 total pixels (GEMM M)
#define MTILES (NPIX / 16)          // 1024
#define NSTATE ((size_t)BATCH * HID * HW)   // one (B,64,16,16) tensor

__device__ __forceinline__ float sigmf(float x) { return 1.f / (1.f + __expf(-x)); }

// WMMA 16x16x32 f16 fragment K-index for (lane, halfIdx j). A: row = lane&15.
// lanes 0-15 hold K {base..base+7, base+16..base+23} with base=0; lanes 16-31 base=8.
__device__ __forceinline__ int kfrag(int lane, int j) {
  int base = (lane >> 4) << 3;
  return (j < 8) ? (base + j) : (8 + base + j);   // j>=8 -> 16 + base + (j-8)
}

// ---------------------------------------------------------------------------
// rec_avg[b][r][m] = mean over pixels of recalled (B,240,16,16); 240 = r*60+m
// ---------------------------------------------------------------------------
__global__ void __launch_bounds__(256) recavg_kernel(const float* __restrict__ rec,
                                                     float* __restrict__ rec_avg) {
  int b = blockIdx.x, tid = threadIdx.x;
  if (tid < RLEN * MG) {
    const float* p = rec + ((size_t)b * 240 + tid) * HW;
    float s = 0.f;
    for (int i = 0; i < HW; ++i) s += p[i];
    rec_avg[(size_t)b * 240 + tid] = s * (1.f / 256.f);
  }
}

// ---------------------------------------------------------------------------
// Pack 3x3 conv weights (256, Cin, 3, 3) -> B-fragments [nt][kt][lane][16] f16
// ---------------------------------------------------------------------------
__global__ void __launch_bounds__(256) pack_w_gates(const float* __restrict__ w,
                                                    int Cin, int Ktiles,
                                                    _Float16* __restrict__ out) {
  int idx = blockIdx.x * 256 + threadIdx.x;
  int total = 16 * Ktiles * 512;
  if (idx >= total) return;
  int j    = idx & 15;
  int lane = (idx >> 4) & 31;
  int kt   = (idx >> 9) % Ktiles;
  int nt   = idx / (Ktiles << 9);
  int n    = nt * 16 + (lane & 15);
  int k    = kt * 32 + kfrag(lane, j);
  int Ktot = Cin * 9;
  float v = 0.f;
  if (k < Ktot) {
    int c = k / 9, rs = k % 9;
    v = w[((size_t)n * Cin + c) * 9 + rs];
  }
  out[idx] = (_Float16)v;
}

// Pack 1x1 weights (N, Ktot) -> B-fragments [nt][kt][lane][16]
__global__ void __launch_bounds__(256) pack_w_1x1(const float* __restrict__ w,
                                                  int Ktot, int Ktiles, int Ntiles,
                                                  _Float16* __restrict__ out) {
  int idx = blockIdx.x * 256 + threadIdx.x;
  int total = Ntiles * Ktiles * 512;
  if (idx >= total) return;
  int j    = idx & 15;
  int lane = (idx >> 4) & 31;
  int kt   = (idx >> 9) % Ktiles;
  int nt   = idx / (Ktiles << 9);
  int n    = nt * 16 + (lane & 15);
  int k    = kt * 32 + kfrag(lane, j);
  float v = (k < Ktot) ? w[(size_t)n * Ktot + k] : 0.f;
  out[idx] = (_Float16)v;
}

// ---------------------------------------------------------------------------
// im2col (3x3, pad 1) of concat(in0[C0], in1[C1]) -> A-fragments
// [mt][kt][lane][16] f16.  Pixel order: m = b*256 + y*16 + x.
// ---------------------------------------------------------------------------
__global__ void __launch_bounds__(256) pack_im2col(const float* __restrict__ in0, int C0,
                                                   const float* __restrict__ in1, int C1,
                                                   int Ktiles, _Float16* __restrict__ out) {
  int idx = blockIdx.x * 256 + threadIdx.x;
  int total = MTILES * Ktiles * 512;
  if (idx >= total) return;
  int j    = idx & 15;
  int lane = (idx >> 4) & 31;
  int kt   = (idx >> 9) % Ktiles;
  int mt   = idx / (Ktiles << 9);
  int m    = mt * 16 + (lane & 15);
  int b = m >> 8, pix = m & 255, y = pix >> 4, x = pix & 15;
  int k    = kt * 32 + kfrag(lane, j);
  int Ktot = (C0 + C1) * 9;
  float v = 0.f;
  if (k < Ktot) {
    int c = k / 9, r = (k % 9) / 3, s = k % 3;
    int iy = y + r - 1, ix = x + s - 1;
    if (iy >= 0 && iy < 16 && ix >= 0 && ix < 16) {
      v = (c < C0) ? in0[(((size_t)b * C0 + c) << 8) + iy * 16 + ix]
                   : in1[(((size_t)b * C1 + (c - C0)) << 8) + iy * 16 + ix];
    }
  }
  out[idx] = (_Float16)v;
}

// 1x1 activation pack of concat(in0[C0], in1[C1]) -> A-fragments (Ktiles fixed 4)
__global__ void __launch_bounds__(256) pack_act_1x1(const float* __restrict__ in0, int C0,
                                                    const float* __restrict__ in1, int C1,
                                                    _Float16* __restrict__ out) {
  int idx = blockIdx.x * 256 + threadIdx.x;
  int total = MTILES * 4 * 512;
  if (idx >= total) return;
  int j    = idx & 15;
  int lane = (idx >> 4) & 31;
  int kt   = (idx >> 9) & 3;
  int mt   = idx >> 11;
  int m    = mt * 16 + (lane & 15);
  int b = m >> 8, pix = m & 255;
  int k    = kt * 32 + kfrag(lane, j);
  float v = 0.f;
  if (k < C0)            v = in0[(((size_t)b * C0 + k) << 8) + pix];
  else if (k < C0 + C1)  v = in1[(((size_t)b * C1 + (k - C0)) << 8) + pix];
  out[idx] = (_Float16)v;
}

// ---------------------------------------------------------------------------
// Per-timestep perception attention: q = leaky_relu(conv1x1(c, wc)+bc),
// attn = softmax(cos(q_avg, rec_avg_r)), mt = sum_r attn_r * rec_feat[b,r].
// One block (256 thr) per batch element.
// ---------------------------------------------------------------------------
__global__ void __launch_bounds__(256) attention_kernel(const float* __restrict__ cprev,
                                                        const float* __restrict__ rec,
                                                        const float* __restrict__ rec_avg,
                                                        const float* __restrict__ wc,
                                                        const float* __restrict__ bc,
                                                        float* __restrict__ mtout) {
  int b = blockIdx.x, tid = threadIdx.x;
  __shared__ float qsum[MG];
  __shared__ float sim[RLEN];
  __shared__ float attn[RLEN];
  if (tid < MG) qsum[tid] = 0.f;
  __syncthreads();
  if (tid < 240) {                 // (m, quarter-of-pixels) workers
    int m = tid % MG, quarter = tid / MG;
    const float* cb = cprev + (size_t)b * HID * HW;
    const float* wrow = wc + m * HID;
    float bias = bc[m];
    float acc = 0.f;
    int p0 = quarter * 64;
    for (int p = p0; p < p0 + 64; ++p) {
      float v = bias;
      for (int ch = 0; ch < HID; ++ch) v += wrow[ch] * cb[ch * HW + p];
      acc += (v > 0.f) ? v : 0.2f * v;          // leaky_relu 0.2
    }
    atomicAdd(&qsum[m], acc);                   // ds_add_f32
  }
  __syncthreads();
  if (tid < RLEN) {
    const float* ra = rec_avg + ((size_t)b * RLEN + tid) * MG;
    float num = 0.f, qn = 0.f, rn = 0.f;
    for (int m = 0; m < MG; ++m) {
      float qa = qsum[m] * (1.f / 256.f);
      num += qa * ra[m]; qn += qa * qa; rn += ra[m] * ra[m];
    }
    float den = fmaxf(sqrtf(qn), 1e-8f) * fmaxf(sqrtf(rn), 1e-8f);
    sim[tid] = num / den;
  }
  __syncthreads();
  if (tid == 0) {
    float mx = fmaxf(fmaxf(sim[0], sim[1]), fmaxf(sim[2], sim[3]));
    float e0 = __expf(sim[0]-mx), e1 = __expf(sim[1]-mx),
          e2 = __expf(sim[2]-mx), e3 = __expf(sim[3]-mx);
    float inv = 1.f / (e0 + e1 + e2 + e3);
    attn[0] = e0*inv; attn[1] = e1*inv; attn[2] = e2*inv; attn[3] = e3*inv;
  }
  __syncthreads();
  const float* rb = rec + (size_t)b * 240 * HW;
  float a0 = attn[0], a1 = attn[1], a2 = attn[2], a3 = attn[3];
  for (int e = tid; e < MG * HW; e += 256) {
    int m = e >> 8, p = e & 255;
    mtout[((size_t)b * MG + m) * HW + p] =
        a0 * rb[(       m) * HW + p] + a1 * rb[( 60 + m) * HW + p] +
        a2 * rb[(120 + m) * HW + p] + a3 * rb[(180 + m) * HW + p];
  }
}

// ---------------------------------------------------------------------------
// Gates GEMM (implicit 3x3 conv): one block per 32-pixel (2 M-tile) slab;
// 8 waves cover all 16 N-tiles, each wave holds a 2x2 accumulator grid so
// every B fragment feeds 2 WMMAs.  A-tiles are DMA'd into a double-buffered
// LDS staging area with GLOBAL_LOAD_ASYNC_TO_LDS_B128; a partial
// s_wait_asynccnt overlaps the copy of chunk i+1 with compute on chunk i.
// Epilogue fuses LSTM cell math via a 32KB LDS gate buffer.
// ---------------------------------------------------------------------------
__global__ void __launch_bounds__(256) gemm_gates(const _Float16* __restrict__ A,
                                                  const _Float16* __restrict__ W,
                                                  const float* __restrict__ bias,
                                                  const float* __restrict__ cprev,
                                                  float* __restrict__ cnew,
                                                  float* __restrict__ osig,
                                                  int Ktiles) {
  int mt0 = blockIdx.x * 2;                        // two M-tiles per block
  int tid = threadIdx.x, wave = tid >> 5, lane = tid & 31;
  __shared__ __align__(16) _Float16 As[2][2][4 * 512]; // [buf][mti][4 ktiles] 16 KB
  __shared__ float gts[32 * 256];                      // 32 pixels x 256 gates, 32 KB

  // LDS byte offset: low 32 bits of the shared aperture address hold the
  // wave-relative LDS offset (ISA 10.2).
  unsigned lbase = (unsigned)(uintptr_t)As;

  int nt0 = wave * 2, nt1 = nt0 + 1;
  const v16h* B0 = (const v16h*)W + (size_t)nt0 * Ktiles * 32 + lane;
  const v16h* B1 = (const v16h*)W + (size_t)nt1 * Ktiles * 32 + lane;
  const char* Ag0 = (const char*)(A + (size_t)(mt0 + 0) * Ktiles * 512) + tid * 16;
  const char* Ag1 = (const char*)(A + (size_t)(mt0 + 1) * Ktiles * 512) + tid * 16;

  // async DMA of one 4-ktile chunk (4 KB per M-tile) into staging buffer buf
  auto stage = [&](int ch, int buf) {
    unsigned l0 = lbase + (unsigned)buf * 8192u + (unsigned)tid * 16u;
    unsigned l1 = l0 + 4096u;
    unsigned long long g0 = (unsigned long long)(uintptr_t)(Ag0 + (size_t)ch * 4096);
    unsigned long long g1 = (unsigned long long)(uintptr_t)(Ag1 + (size_t)ch * 4096);
    asm volatile("global_load_async_to_lds_b128 %0, %1, off" :: "v"(l0), "v"(g0) : "memory");
    asm volatile("global_load_async_to_lds_b128 %0, %1, off" :: "v"(l1), "v"(g1) : "memory");
  };

  int nchunks = Ktiles >> 2;                       // Ktiles is a multiple of 4
  stage(0, 0);

  v8f acc00 = {}, acc01 = {}, acc10 = {}, acc11 = {};
  for (int ch = 0; ch < nchunks; ++ch) {
    int buf = ch & 1;
    if (ch + 1 < nchunks) {
      stage(ch + 1, buf ^ 1);                      // overlap next copy w/ compute
      asm volatile("s_wait_asynccnt 2" ::: "memory");  // chunk ch done (in-order)
    } else {
      asm volatile("s_wait_asynccnt 0" ::: "memory");
    }
    __syncthreads();
#pragma unroll
    for (int kl = 0; kl < 4; ++kl) {
      int kt = ch * 4 + kl;
      v16h b0 = B0[kt * 32];
      v16h b1 = B1[kt * 32];
      v16h a0 = *(const v16h*)&As[buf][0][(kl * 32 + lane) * 16];  // ds_load
      v16h a1 = *(const v16h*)&As[buf][1][(kl * 32 + lane) * 16];
      if (kl == 3 && ch + 1 < nchunks) {                           // prefetch B
        __builtin_prefetch((const void*)(B0 + (kt + 1) * 32), 0, 1);
        __builtin_prefetch((const void*)(B1 + (kt + 1) * 32), 0, 1);
      }
      acc00 = __builtin_amdgcn_wmma_f32_16x16x32_f16(false, a0, false, b0, (short)0, acc00, false, false);
      acc01 = __builtin_amdgcn_wmma_f32_16x16x32_f16(false, a0, false, b1, (short)0, acc01, false, false);
      acc10 = __builtin_amdgcn_wmma_f32_16x16x32_f16(false, a1, false, b0, (short)0, acc10, false, false);
      acc11 = __builtin_amdgcn_wmma_f32_16x16x32_f16(false, a1, false, b1, (short)0, acc11, false, false);
    }
    __syncthreads();                       // chunk fully consumed before reuse
  }

  int nlo = lane & 15, mhi = (lane >> 4) << 3;
  int n0 = nt0 * 16 + nlo, n1 = nt1 * 16 + nlo;
  float bb0 = bias[n0], bb1 = bias[n1];
#pragma unroll
  for (int r = 0; r < 8; ++r) {
    int m = r + mhi;
    gts[(     m) * 256 + n0] = acc00[r] + bb0;
    gts[(     m) * 256 + n1] = acc01[r] + bb1;
    gts[(16 + m) * 256 + n0] = acc10[r] + bb0;
    gts[(16 + m) * 256 + n1] = acc11[r] + bb1;
  }
  __syncthreads();

  for (int e = tid; e < 32 * HID; e += 256) {    // 32 pixels x 64 cell channels
    int m = e >> 6, chn = e & 63;
    int gm = mt0 * 16 + m;
    int b = gm >> 8, pix = gm & 255;
    float ig = gts[m * 256 + chn];
    float fg = gts[m * 256 +  64 + chn];
    float gg = gts[m * 256 + 128 + chn];
    float og = gts[m * 256 + 192 + chn];
    size_t off = (((size_t)b * HID + chn) << 8) + pix;
    float cn = sigmf(fg) * cprev[off] + sigmf(ig) * tanhf(gg);
    cnew[off] = cn;
    osig[off] = sigmf(og);
  }
}

// ---------------------------------------------------------------------------
// Output 1x1 GEMM: h = sig(o) * tanh(conv1x1(concat(c_new, mt), wf)+bf).
// Ktiles fixed 4 (K=124 padded to 128). Wave -> one 16x16 tile; grid 512.
// ---------------------------------------------------------------------------
__global__ void __launch_bounds__(256) gemm_hnew(const _Float16* __restrict__ A,
                                                 const _Float16* __restrict__ Wf,
                                                 const float* __restrict__ bf,
                                                 const float* __restrict__ osig,
                                                 float* __restrict__ h,
                                                 float* __restrict__ out_opt) {
  int tid = threadIdx.x, wave = tid >> 5, lane = tid & 31;
  int mt = blockIdx.x * 2 + (wave >> 2);
  int nt = wave & 3;
  const v16h* Ap = (const v16h*)A  + (size_t)mt * 4 * 32 + lane;
  const v16h* Bp = (const v16h*)Wf + (size_t)nt * 4 * 32 + lane;
  v8f acc = {};
#pragma unroll
  for (int kt = 0; kt < 4; ++kt) {
    v16h a = Ap[kt * 32];
    v16h b = Bp[kt * 32];
    acc = __builtin_amdgcn_wmma_f32_16x16x32_f16(false, a, false, b, (short)0, acc, false, false);
  }
  int nlo = lane & 15, mhi = (lane >> 4) << 3;
  int n = nt * 16 + nlo;
  float bb = bf[n];
#pragma unroll
  for (int r = 0; r < 8; ++r) {
    int gm = mt * 16 + r + mhi;
    int b = gm >> 8, pix = gm & 255;
    size_t off = (((size_t)b * HID + n) << 8) + pix;
    float hv = osig[off] * tanhf(acc[r] + bb);
    h[off] = hv;
    if (out_opt) out_opt[off] = hv;
  }
}

// ---------------------------------------------------------------------------
// Host orchestration
// ---------------------------------------------------------------------------
extern "C" void kernel_launch(void* const* d_in, const int* in_sizes, int n_in,
                              void* d_out, int out_size, void* d_ws, size_t ws_size,
                              hipStream_t stream) {
  const float* x    = (const float*)d_in[0];   // (10,64,8,16,16)
  const float* rec  = (const float*)d_in[1];   // (64,240,16,16)
  const float* w0   = (const float*)d_in[2];   // (256,72,3,3)
  const float* b0   = (const float*)d_in[3];
  const float* wc0  = (const float*)d_in[4];   // (60,64)
  const float* bc0  = (const float*)d_in[5];
  const float* wf0  = (const float*)d_in[6];   // (64,124)
  const float* bf0  = (const float*)d_in[7];
  const float* w1   = (const float*)d_in[8];   // (256,128,3,3)
  const float* b1   = (const float*)d_in[9];
  const float* wc1  = (const float*)d_in[10];
  const float* bc1  = (const float*)d_in[11];
  const float* wf1  = (const float*)d_in[12];
  const float* bf1  = (const float*)d_in[13];
  float* out = (float*)d_out;

  const int KT0 = 24;   // ceil(72*9/32)=21, padded to multiple of 4 for LDS chunks
  const int KT1 = 36;   // 128*9/32 (already multiple of 4)
  const size_t SB = NSTATE * sizeof(float);    // 4 MB state tensor

  // workspace carve-up (256B aligned)
  char* ws = (char*)d_ws;
  size_t off = 0;
  auto carve = [&](size_t bytes) {
    void* p = ws + off;
    off = (off + bytes + 255) & ~(size_t)255;
    return p;
  };
  float* c0a   = (float*)carve(SB);
  float* c0b   = (float*)carve(SB);
  float* c1a   = (float*)carve(SB);
  float* c1b   = (float*)carve(SB);
  float* h0    = (float*)carve(SB);
  float* h1    = (float*)carve(SB);
  float* osig  = (float*)carve(SB);
  float* mtbuf = (float*)carve((size_t)BATCH * MG * HW * sizeof(float));
  float* ravg  = (float*)carve((size_t)BATCH * RLEN * MG * sizeof(float));
  _Float16* pw0  = (_Float16*)carve((size_t)16 * KT0 * 512 * sizeof(_Float16));
  _Float16* pw1  = (_Float16*)carve((size_t)16 * KT1 * 512 * sizeof(_Float16));
  _Float16* pwf0 = (_Float16*)carve((size_t)4 * 4 * 512 * sizeof(_Float16));
  _Float16* pwf1 = (_Float16*)carve((size_t)4 * 4 * 512 * sizeof(_Float16));
  _Float16* pA   = (_Float16*)carve((size_t)MTILES * KT1 * 512 * sizeof(_Float16));
  _Float16* pA1  = (_Float16*)carve((size_t)MTILES * 4 * 512 * sizeof(_Float16));

  float* c0[2] = {c0a, c0b};
  float* c1[2] = {c1a, c1b};

  // zero initial carries (poisoned workspace otherwise)
  hipMemsetAsync(c0a, 0, SB, stream);
  hipMemsetAsync(c1a, 0, SB, stream);
  hipMemsetAsync(h0, 0, SB, stream);
  hipMemsetAsync(h1, 0, SB, stream);

  recavg_kernel<<<BATCH, 256, 0, stream>>>(rec, ravg);
  pack_w_gates<<<(16 * KT0 * 512 + 255) / 256, 256, 0, stream>>>(w0, 72, KT0, pw0);
  pack_w_gates<<<(16 * KT1 * 512 + 255) / 256, 256, 0, stream>>>(w1, 128, KT1, pw1);
  pack_w_1x1<<<(4 * 4 * 512 + 255) / 256, 256, 0, stream>>>(wf0, 124, 4, 4, pwf0);
  pack_w_1x1<<<(4 * 4 * 512 + 255) / 256, 256, 0, stream>>>(wf1, 124, 4, 4, pwf1);

  const int g_im0 = (MTILES * KT0 * 512) / 256;
  const int g_im1 = (MTILES * KT1 * 512) / 256;
  const int g_a1  = (MTILES * 4 * 512) / 256;

  for (int t = 0; t < SEQ; ++t) {
    const float* xt = x + (size_t)t * BATCH * IN_CH * HW;

    // ---- layer 0: input = concat(x_t, h0_prev) ----
    float *c0p = c0[t & 1], *c0n = c0[(t + 1) & 1];
    pack_im2col<<<g_im0, 256, 0, stream>>>(xt, IN_CH, h0, HID, KT0, pA);
    attention_kernel<<<BATCH, 256, 0, stream>>>(c0p, rec, ravg, wc0, bc0, mtbuf);
    gemm_gates<<<MTILES / 2, 256, 0, stream>>>(pA, pw0, b0, c0p, c0n, osig, KT0);
    pack_act_1x1<<<g_a1, 256, 0, stream>>>(c0n, HID, mtbuf, MG, pA1);
    gemm_hnew<<<MTILES / 2, 256, 0, stream>>>(pA1, pwf0, bf0, osig, h0, nullptr);

    // ---- layer 1: input = concat(h0_now, h1_prev) ----
    float *c1p = c1[t & 1], *c1n = c1[(t + 1) & 1];
    pack_im2col<<<g_im1, 256, 0, stream>>>(h0, HID, h1, HID, KT1, pA);
    attention_kernel<<<BATCH, 256, 0, stream>>>(c1p, rec, ravg, wc1, bc1, mtbuf);
    gemm_gates<<<MTILES / 2, 256, 0, stream>>>(pA, pw1, b1, c1p, c1n, osig, KT1);
    pack_act_1x1<<<g_a1, 256, 0, stream>>>(c1n, HID, mtbuf, MG, pA1);
    gemm_hnew<<<MTILES / 2, 256, 0, stream>>>(pA1, pwf1, bf1, osig, h1,
                                              out + (size_t)t * NSTATE);
  }

  // h_last, c_last (layer 1). Final c buffer index = SEQ & 1 = 0 -> c1[0].
  hipMemcpyAsync(out + (size_t)SEQ * NSTATE, h1, SB, hipMemcpyDeviceToDevice, stream);
  hipMemcpyAsync(out + (size_t)(SEQ + 1) * NSTATE, c1[SEQ & 1], SB,
                 hipMemcpyDeviceToDevice, stream);
}
